// CardaicCircleNet_78675210928495
// MI455X (gfx1250) — compile-verified, matrix operand order checked by reading, not run on hardware
//
#include <hip/hip_runtime.h>
#include <hip/hip_bf16.h>

// ---------------------------------------------------------------------------
// CardaicCircleNet forward on gfx1250 (MI455X).
// Conv layers (Cin multiple of 32) run as tap-decomposed implicit GEMM using
// v_wmma_f32_16x16x32_f16, NHWC f16 activations, [tap][Cout][Cin] f16 weights.
// Each wave owns a 16(cout) x 64(pixel) tile: A-frag reused across 4 B-frags.
// All conv shapes are template-specialized (compile-time strides/trip counts).
// ---------------------------------------------------------------------------

#define KB   8      // batch
#define KIMG 128
#define KN   32     // circle points
#define KV   33     // nodes per circle (+center)
#define KCP0 16     // N/2

typedef __attribute__((ext_vector_type(16))) _Float16 v16h;
typedef __attribute__((ext_vector_type(8)))  _Float16 v8h;
typedef __attribute__((ext_vector_type(8)))  float    v8f;

static __device__ __forceinline__ v16h comb16(v8h lo, v8h hi) {
  v16h r;
#pragma unroll
  for (int i = 0; i < 8; ++i) { r[i] = lo[i]; r[i + 8] = hi[i]; }
  return r;
}

// ---------------------------------------------------------------------------
// 1) per-image min/max normalize -> f16 plane
// ---------------------------------------------------------------------------
__global__ void k_norm(const float* __restrict__ img, _Float16* __restrict__ out) {
  int b = blockIdx.x;
  int t = threadIdx.x;
  const float* p = img + (size_t)b * KIMG * KIMG;
  float mn = 1e30f, mx = -1e30f;
  for (int i = t; i < KIMG * KIMG; i += blockDim.x) {
    float v = p[i];
    mn = fminf(mn, v); mx = fmaxf(mx, v);
  }
  __shared__ float smn[256], smx[256];
  smn[t] = mn; smx[t] = mx;
  __syncthreads();
  for (int s = 128; s > 0; s >>= 1) {
    if (t < s) { smn[t] = fminf(smn[t], smn[t + s]); smx[t] = fmaxf(smx[t], smx[t + s]); }
    __syncthreads();
  }
  float lo = smn[0], hi = smx[0];
  float inv = 1.0f / (hi - lo + 0.01f);
  for (int i = t; i < KIMG * KIMG; i += blockDim.x)
    out[(size_t)b * KIMG * KIMG + i] = (_Float16)((p[i] - lo) * inv);
}

// ---------------------------------------------------------------------------
// 2) first conv (Cin=1, 5x5, relu) scalar -> NHWC f16 (B,128,128,64)
// ---------------------------------------------------------------------------
__global__ void k_conv1(const _Float16* __restrict__ x, const float* __restrict__ w,
                        const float* __restrict__ bias, _Float16* __restrict__ out) {
  int idx = blockIdx.x * blockDim.x + threadIdx.x;
  int total = KB * KIMG * KIMG;
  if (idx >= total) return;
  int px = idx % KIMG, py = (idx / KIMG) % KIMG, b = idx / (KIMG * KIMG);
  float in[25];
#pragma unroll
  for (int t = 0; t < 25; ++t) {
    int sy = py + t / 5 - 2, sx = px + t % 5 - 2;
    bool inb = (sy >= 0 && sy < KIMG && sx >= 0 && sx < KIMG);
    in[t] = inb ? (float)x[((size_t)b * KIMG + sy) * KIMG + sx] : 0.f;
  }
  _Float16* o = out + (size_t)idx * 64;
  for (int co = 0; co < 64; ++co) {
    float acc = bias[co];
    const float* wr = w + co * 25;
#pragma unroll
    for (int t = 0; t < 25; ++t) acc += in[t] * wr[t];
    o[co] = (_Float16)fmaxf(acc, 0.f);
  }
}

// ---------------------------------------------------------------------------
// 3) NHWC 2x2 max pool
// ---------------------------------------------------------------------------
__global__ void k_pool(const _Float16* __restrict__ in, _Float16* __restrict__ out,
                       int H, int W, int C) {
  int oH = H / 2, oW = W / 2;
  long long total = (long long)KB * oH * oW * C;
  long long idx = (long long)blockIdx.x * blockDim.x + threadIdx.x;
  if (idx >= total) return;
  int c = idx % C;
  int ox = (idx / C) % oW;
  int oy = (idx / ((long long)C * oW)) % oH;
  int b = idx / ((long long)C * oW * oH);
  const _Float16* p = in + (((size_t)b * H + 2 * oy) * W + 2 * ox) * C + c;
  float m = (float)p[0];
  m = fmaxf(m, (float)p[C]);
  m = fmaxf(m, (float)p[(size_t)W * C]);
  m = fmaxf(m, (float)p[(size_t)W * C + C]);
  out[idx] = (_Float16)m;
}

// ---------------------------------------------------------------------------
// 4) weight convert f32 OIHW -> f16 [tap][CoutPad][Cin] (zero-pad extra couts)
// ---------------------------------------------------------------------------
__global__ void k_wconv(const float* __restrict__ w, _Float16* __restrict__ out,
                        int Cout, int CoutPad, int Cin, int taps) {
  long long total = (long long)taps * CoutPad * Cin;
  long long idx = (long long)blockIdx.x * blockDim.x + threadIdx.x;
  if (idx >= total) return;
  int i = idx % Cin;
  int o = (idx / Cin) % CoutPad;
  int t = idx / ((long long)Cin * CoutPad);
  out[idx] = (o < Cout) ? (_Float16)w[((size_t)o * Cin + i) * taps + t] : (_Float16)0.f;
}

// ---------------------------------------------------------------------------
// 5) implicit-GEMM conv via WMMA, fully shape-specialized.
//    Each wave: 16(cout) x (16*PIXT)(pixel) tile; A-fragment reused PIXT times.
//    ACT: 0=none 1=relu 2=tanh
// ---------------------------------------------------------------------------
template <int H, int W, int CIN, int COUTP, int COUT, int TAPS, int ACT, int PIXT>
__global__ void __launch_bounds__(256)
k_conv_wmma(const _Float16* __restrict__ in, const _Float16* __restrict__ w16,
            const float* __restrict__ bias, _Float16* __restrict__ out) {
  constexpr int NPG = (H * W) / (16 * PIXT);   // pixel groups
  constexpr int NCT = COUTP / 16;              // cout tiles
  constexpr int TOT = KB * NCT * NPG;          // waves
  int wid  = (blockIdx.x * blockDim.x + threadIdx.x) >> 5;
  if (wid >= TOT) return;                      // wave-uniform: EXEC stays all-ones
  int lane = threadIdx.x & 31;
  int m16  = lane & 15;
  int half = lane >> 4;
  int pt = wid % NPG;
  int ct = (wid / NPG) % NCT;
  int b  = wid / (NPG * NCT);
  int pix0 = pt * (16 * PIXT) + m16;           // this lane's column in group 0

  v8f acc[PIXT];
#pragma unroll
  for (int u = 0; u < PIXT; ++u) acc[u] = (v8f){};

  const _Float16* arow = w16 + (size_t)(ct * 16 + m16) * CIN;  // A-matrix row
  constexpr int KW  = (TAPS == 25) ? 5 : 1;
  constexpr int PAD = (TAPS == 25) ? 2 : 0;

  for (int tap = 0; tap < TAPS; ++tap) {
    int dy = tap / KW - PAD, dx = tap % KW - PAD;
    const _Float16* brow[PIXT];
    bool inb[PIXT];
#pragma unroll
    for (int u = 0; u < PIXT; ++u) {
      int pix = pix0 + u * 16;
      int py = pix / W, px = pix % W;          // W is power of two -> shifts
      int sy = py + dy, sx = px + dx;
      inb[u] = (sy >= 0 && sy < H && sx >= 0 && sx < W);
      brow[u] = in + (((size_t)b * H + sy) * W + sx) * CIN;
    }
    const _Float16* ar = arow + (size_t)tap * COUTP * CIN;
#pragma unroll
    for (int k0 = 0; k0 < CIN; k0 += 32) {
      // A frag: lane holds row m16, K = k0+half*8+0..7 / k0+16+half*8+0..7
      v8h alo = *(const v8h*)(ar + k0 + half * 8);
      v8h ahi = *(const v8h*)(ar + k0 + 16 + half * 8);
      v16h Af = comb16(alo, ahi);
#pragma unroll
      for (int u = 0; u < PIXT; ++u) {
        v8h blo = {}, bhi = {};
        if (inb[u]) {
          blo = *(const v8h*)(brow[u] + k0 + half * 8);
          bhi = *(const v8h*)(brow[u] + k0 + 16 + half * 8);
        }
        v16h Bf = comb16(blo, bhi);
        acc[u] = __builtin_amdgcn_wmma_f32_16x16x32_f16(
            false, Af, false, Bf, (short)0, acc[u], false, false);
      }
    }
  }
  // epilogue: lane owns couts (ct*16 + half*8 + j), contiguous in NHWC
  int cbase = ct * 16 + half * 8;
#pragma unroll
  for (int u = 0; u < PIXT; ++u) {
    int pix = pix0 + u * 16;
    int py = pix / W, px = pix % W;
    v8h o;
#pragma unroll
    for (int j = 0; j < 8; ++j) {
      int c = cbase + j;
      float v = acc[u][j] + ((c < COUT) ? bias[c] : 0.f);
      if (ACT == 1) v = fmaxf(v, 0.f);
      else if (ACT == 2) v = tanhf(v);
      o[j] = (_Float16)v;
    }
    *(v8h*)(out + (((size_t)b * H + py) * W + px) * COUTP + cbase) = o;
  }
}

// ---------------------------------------------------------------------------
// 6) bilinear 2x upsample NHWC f16, writing into (possibly wider) dst channels
// ---------------------------------------------------------------------------
__global__ void k_resize2x(const _Float16* __restrict__ in, _Float16* __restrict__ dst,
                           int sH, int sW, int C, int dstC, int coff) {
  int dH = 2 * sH, dW = 2 * sW;
  long long total = (long long)KB * dH * dW * C;
  long long idx = (long long)blockIdx.x * blockDim.x + threadIdx.x;
  if (idx >= total) return;
  int c = idx % C;
  int dx = (idx / C) % dW;
  int dy = (idx / ((long long)C * dW)) % dH;
  int b = idx / ((long long)C * dW * dH);
  float sx = (dx + 0.5f) * 0.5f - 0.5f;
  float sy = (dy + 0.5f) * 0.5f - 0.5f;
  float fx = floorf(sx), fy = floorf(sy);
  int x0 = (int)fx, y0 = (int)fy;
  float xw = sx - fx, yw = sy - fy;
  int x0c = min(max(x0, 0), sW - 1), x1c = min(max(x0 + 1, 0), sW - 1);
  int y0c = min(max(y0, 0), sH - 1), y1c = min(max(y0 + 1, 0), sH - 1);
  const _Float16* base = in + (size_t)b * sH * sW * C + c;
  float v00 = (float)base[((size_t)y0c * sW + x0c) * C];
  float v01 = (float)base[((size_t)y0c * sW + x1c) * C];
  float v10 = (float)base[((size_t)y1c * sW + x0c) * C];
  float v11 = (float)base[((size_t)y1c * sW + x1c) * C];
  float v = v00 * (1 - xw) * (1 - yw) + v01 * xw * (1 - yw)
          + v10 * (1 - xw) * yw + v11 * xw * yw;
  dst[(((size_t)b * dH + dy) * dW + dx) * dstC + coff + c] = (_Float16)v;
}

// ---------------------------------------------------------------------------
// 7) channel-block copy into concat buffer
// ---------------------------------------------------------------------------
__global__ void k_copyc(const _Float16* __restrict__ src, _Float16* __restrict__ dst,
                        int H, int W, int Cs, int dstC, int coff) {
  long long total = (long long)KB * H * W * Cs;
  long long idx = (long long)blockIdx.x * blockDim.x + threadIdx.x;
  if (idx >= total) return;
  int c = idx % Cs;
  long long p = idx / Cs;       // b*H*W + y*W + x
  dst[p * dstC + coff + c] = src[idx];
}

// ---------------------------------------------------------------------------
// 8) small dense layers. remap!=0: input is g2 NHWC (B,4,4,128) flattened as NCHW
// ---------------------------------------------------------------------------
__global__ void k_linear(const _Float16* __restrict__ inH, const float* __restrict__ inF,
                         const float* __restrict__ Wt, const float* __restrict__ bias,
                         float* __restrict__ out, int K, int J, int act, int remap) {
  int idx = blockIdx.x * blockDim.x + threadIdx.x;
  if (idx >= KB * J) return;
  int j = idx % J, b = idx / J;
  float acc = bias[j];
  for (int k = 0; k < K; ++k) {
    float v;
    if (remap) {
      int c = k >> 4, p = k & 15, y = p >> 2, x = p & 3;
      v = (float)inH[(((size_t)b * 4 + y) * 4 + x) * 128 + c];
    } else {
      v = inF[(size_t)b * K + k];
    }
    acc += v * Wt[(size_t)k * J + j];
  }
  if (act == 1) acc = fmaxf(acc, 0.f);
  else if (act == 2) acc = tanhf(acc);
  out[(size_t)b * J + j] = acc;
}

// ---------------------------------------------------------------------------
// 9) build circle nodes and apply affine (h @ A, translation column discarded)
// ---------------------------------------------------------------------------
__global__ void k_nodes(const float* __restrict__ aff, float* __restrict__ n1,
                        float* __restrict__ n2) {
  int idx = blockIdx.x * blockDim.x + threadIdx.x;
  if (idx >= KB * KV) return;
  int i = idx % KV, b = idx / KV;
  const float* a = aff + (size_t)b * 6;
  float bx1, by1, bx2, by2;
  if (i < KN) {
    float th = 6.28318530717958647692f * (float)i / (float)KN;
    float cx = cosf(th), cy = sinf(th);
    bx1 = 0.5f * cx;        by1 = 0.5f * cy;
    bx2 = 0.3f * cx + 0.1f; by2 = 0.3f * cy;
  } else {
    bx1 = 0.f; by1 = 0.f; bx2 = 0.1f; by2 = 0.f;
  }
  n1[2 * idx]     = bx1 * a[0] + by1 * a[3];
  n1[2 * idx + 1] = bx1 * a[1] + by1 * a[4];
  n2[2 * idx]     = bx2 * a[0] + by2 * a[3];
  n2[2 * idx + 1] = bx2 * a[1] + by2 * a[4];
}

// ---------------------------------------------------------------------------
// 10) soft rasterizer: one thread per pixel, loop over 32 triangle fans
// ---------------------------------------------------------------------------
__global__ void k_render(const float* __restrict__ nodes, float* __restrict__ out,
                         int chan) {
  long long idx = (long long)blockIdx.x * blockDim.x + threadIdx.x;
  if (idx >= (long long)KB * KIMG * KIMG) return;
  int p = idx % (KIMG * KIMG);
  int b = idx / (KIMG * KIMG);
  int ix = p % KIMG, iy = p / KIMG;
  float Px = (ix + 0.5f) * (2.f / KIMG) - 1.f;
  float Py = 1.f - (iy + 0.5f) * (2.f / KIMG);
  const float* nd = nodes + (size_t)b * KV * 2;
  float cx = nd[2 * KN], cy = nd[2 * KN + 1];
  float m = 0.f;
  for (int f = 0; f < KN; ++f) {
    int f1 = (f + 1 == KN) ? 0 : f + 1;
    float v0x = nd[2 * f], v0y = nd[2 * f + 1];
    float v1x = nd[2 * f1], v1y = nd[2 * f1 + 1];
    float a2 = (v1x - v0x) * (cy - v0y) - (v1y - v0y) * (cx - v0x);
    float s = (a2 >= 0.f) ? 1.f : -1.f;
    float ex = v1x - v0x, ey = v1y - v0y;
    float d0 = (ex * (Py - v0y) - ey * (Px - v0x)) / (sqrtf(ex * ex + ey * ey) + 1e-8f);
    ex = cx - v1x; ey = cy - v1y;
    float d1 = (ex * (Py - v1y) - ey * (Px - v1x)) / (sqrtf(ex * ex + ey * ey) + 1e-8f);
    ex = v0x - cx; ey = v0y - cy;
    float d2 = (ex * (Py - cy) - ey * (Px - cx)) / (sqrtf(ex * ex + ey * ey) + 1e-8f);
    float dmin = fminf(fminf(s * d0, s * d1), s * d2);
    float soft = 1.f / (1.f + expf(-128.f * dmin));
    m = fmaxf(m, soft);
  }
  out[((size_t)b * 4 + chan) * KIMG * KIMG + p] = m;
}

// ---------------------------------------------------------------------------
// 11) disp upsample (B,64,64,16 f16 NHWC, valid c<4) -> (B,4,128,128) f32
// ---------------------------------------------------------------------------
__global__ void k_resize_disp(const _Float16* __restrict__ in, float* __restrict__ out) {
  long long total = (long long)KB * 4 * KIMG * KIMG;
  long long idx = (long long)blockIdx.x * blockDim.x + threadIdx.x;
  if (idx >= total) return;
  int x = idx % KIMG;
  int y = (idx / KIMG) % KIMG;
  int c = (idx / (KIMG * KIMG)) % 4;
  int b = idx / (4 * KIMG * KIMG);
  float sx = (x + 0.5f) * 0.5f - 0.5f;
  float sy = (y + 0.5f) * 0.5f - 0.5f;
  float fx = floorf(sx), fy = floorf(sy);
  int x0 = (int)fx, y0 = (int)fy;
  float xw = sx - fx, yw = sy - fy;
  int x0c = min(max(x0, 0), 63), x1c = min(max(x0 + 1, 0), 63);
  int y0c = min(max(y0, 0), 63), y1c = min(max(y0 + 1, 0), 63);
  const _Float16* base = in + (size_t)b * 64 * 64 * 16 + c;
  float v00 = (float)base[((size_t)y0c * 64 + x0c) * 16];
  float v01 = (float)base[((size_t)y0c * 64 + x1c) * 16];
  float v10 = (float)base[((size_t)y1c * 64 + x0c) * 16];
  float v11 = (float)base[((size_t)y1c * 64 + x1c) * 16];
  out[idx] = v00 * (1 - xw) * (1 - yw) + v01 * xw * (1 - yw)
           + v10 * (1 - xw) * yw + v11 * xw * yw;
}

// ---------------------------------------------------------------------------
// 12) one deformation iteration (grid-sample update of n1, n2)
// ---------------------------------------------------------------------------
static __device__ __forceinline__ float gsample(const float* pl, float px, float py) {
  float x = (px + 1.f) * (KIMG * 0.5f) - 0.5f;
  float y = (py + 1.f) * (KIMG * 0.5f) - 0.5f;
  float fx = floorf(x), fy = floorf(y);
  int x0 = (int)fx, y0 = (int)fy;
  float xw = x - fx, yw = y - fy;
  float v00 = (x0 >= 0 && x0 < KIMG && y0 >= 0 && y0 < KIMG) ? pl[y0 * KIMG + x0] : 0.f;
  float v01 = (x0+1 >= 0 && x0+1 < KIMG && y0 >= 0 && y0 < KIMG) ? pl[y0 * KIMG + x0 + 1] : 0.f;
  float v10 = (x0 >= 0 && x0 < KIMG && y0+1 >= 0 && y0+1 < KIMG) ? pl[(y0+1) * KIMG + x0] : 0.f;
  float v11 = (x0+1 >= 0 && x0+1 < KIMG && y0+1 >= 0 && y0+1 < KIMG) ? pl[(y0+1) * KIMG + x0 + 1] : 0.f;
  return v00 * (1 - xw) * (1 - yw) + v01 * xw * (1 - yw)
       + v10 * (1 - xw) * yw + v11 * xw * yw;
}

__global__ void k_iter(const float* __restrict__ disp, float* __restrict__ n1,
                       float* __restrict__ n2) {
  int idx = blockIdx.x * blockDim.x + threadIdx.x;
  if (idx >= KB * KV) return;
  int i = idx % KV, b = idx / KV;
  const float* d0 = disp + ((size_t)b * 4 + 0) * KIMG * KIMG;
  const float* d1 = disp + ((size_t)b * 4 + 1) * KIMG * KIMG;
  const float* d2 = disp + ((size_t)b * 4 + 2) * KIMG * KIMG;
  const float* d3 = disp + ((size_t)b * 4 + 3) * KIMG * KIMG;
  // n1 update
  float gx = n1[2 * idx], gy = -n1[2 * idx + 1];
  float ux = gsample(d0, gx, gy), uy = gsample(d1, gx, gy);
  n1[2 * idx] = gx + ux;
  n1[2 * idx + 1] = -(gy + uy);
  // n2 update
  gx = n2[2 * idx]; gy = -n2[2 * idx + 1];
  float p0x = gsample(d0, gx, gy), p0y = gsample(d1, gx, gy);
  float p2x = gsample(d2, gx, gy), p2y = gsample(d3, gx, gy);
  float w2 = (i <= KCP0) ? 1.f : 0.f;
  float w0 = ((i >= KCP0) ? 1.f : 0.f) + ((i == KV - 1) ? 1.f : 0.f);
  n2[2 * idx] = gx + w2 * p2x + w0 * p0x;
  n2[2 * idx + 1] = -(gy + w2 * p2y + w0 * p0y);
}

// ---------------------------------------------------------------------------
// host orchestration
// ---------------------------------------------------------------------------
extern "C" void kernel_launch(void* const* d_in, const int* in_sizes, int n_in,
                              void* d_out, int out_size, void* d_ws, size_t ws_size,
                              hipStream_t stream) {
  const float* img = (const float*)d_in[0];
  const float* ew1 = (const float*)d_in[1];  const float* eb1 = (const float*)d_in[2];
  const float* ew2 = (const float*)d_in[3];  const float* eb2 = (const float*)d_in[4];
  const float* ew3 = (const float*)d_in[5];  const float* eb3 = (const float*)d_in[6];
  const float* ew4 = (const float*)d_in[7];  const float* eb4 = (const float*)d_in[8];
  const float* cw1 = (const float*)d_in[9];  const float* cb1 = (const float*)d_in[10];
  const float* cw2 = (const float*)d_in[11]; const float* cb2 = (const float*)d_in[12];
  const float* lw1 = (const float*)d_in[13]; const float* lb1 = (const float*)d_in[14];
  const float* lw2 = (const float*)d_in[15]; const float* lb2 = (const float*)d_in[16];
  const float* lw3 = (const float*)d_in[17]; const float* lb3 = (const float*)d_in[18];
  const float* dw1 = (const float*)d_in[19]; const float* db1 = (const float*)d_in[20];
  const float* dw2 = (const float*)d_in[21]; const float* db2 = (const float*)d_in[22];
  const float* dw3 = (const float*)d_in[23]; const float* db3 = (const float*)d_in[24];
  const float* dwo = (const float*)d_in[25]; const float* dbo = (const float*)d_in[26];
  float* out = (float*)d_out;

  size_t cur = 0;
  auto alloc = [&](size_t bytes) -> char* {
    cur = (cur + 255) & ~(size_t)255;
    char* p = (char*)d_ws + cur;
    cur += bytes;
    return p;
  };
  auto h16 = [&](size_t n) { return (_Float16*)alloc(n * sizeof(_Float16)); };
  auto f32 = [&](size_t n) { return (float*)alloc(n * sizeof(float)); };

  // activations (NHWC f16)
  _Float16* xn  = h16((size_t)KB * 128 * 128);
  _Float16* c1  = h16((size_t)KB * 128 * 128 * 64);
  _Float16* f1  = h16((size_t)KB * 64 * 64 * 64);
  _Float16* c2  = h16((size_t)KB * 64 * 64 * 128);
  _Float16* f2  = h16((size_t)KB * 32 * 32 * 128);
  _Float16* c3  = h16((size_t)KB * 32 * 32 * 256);
  _Float16* f3  = h16((size_t)KB * 16 * 16 * 256);
  _Float16* c4  = h16((size_t)KB * 16 * 16 * 512);
  _Float16* f4  = h16((size_t)KB * 8 * 8 * 512);
  _Float16* g1  = h16((size_t)KB * 8 * 8 * 256);
  _Float16* g1p = h16((size_t)KB * 4 * 4 * 256);
  _Float16* g2  = h16((size_t)KB * 4 * 4 * 128);
  _Float16* cat1 = h16((size_t)KB * 16 * 16 * 768);
  _Float16* d1o  = h16((size_t)KB * 16 * 16 * 256);
  _Float16* cat2 = h16((size_t)KB * 32 * 32 * 384);
  _Float16* d2o  = h16((size_t)KB * 32 * 32 * 128);
  _Float16* cat3 = h16((size_t)KB * 64 * 64 * 192);
  _Float16* d3o  = h16((size_t)KB * 64 * 64 * 64);
  _Float16* d4o  = h16((size_t)KB * 64 * 64 * 16);
  float* disp = f32((size_t)KB * 4 * 128 * 128);
  float* a1 = f32((size_t)KB * 400);
  float* a2 = f32((size_t)KB * 200);
  float* aff = f32((size_t)KB * 6);
  float* n1 = f32((size_t)KB * KV * 2);
  float* n2 = f32((size_t)KB * KV * 2);
  // f16 weights [tap][CoutPad][Cin]
  _Float16* w2f  = h16((size_t)25 * 128 * 64);
  _Float16* w3f  = h16((size_t)25 * 256 * 128);
  _Float16* w4f  = h16((size_t)25 * 512 * 256);
  _Float16* wc1f = h16((size_t)25 * 256 * 512);
  _Float16* wc2f = h16((size_t)1 * 128 * 256);
  _Float16* wd1f = h16((size_t)25 * 256 * 768);
  _Float16* wd2f = h16((size_t)25 * 128 * 384);
  _Float16* wd3f = h16((size_t)25 * 64 * 192);
  _Float16* wdof = h16((size_t)25 * 16 * 64);

  auto grid1 = [](long long n) { return dim3((unsigned)((n + 255) / 256)); };
  auto wconv = [&](const float* w, _Float16* o, int Cout, int CoutPad, int Cin, int taps) {
    long long n = (long long)taps * CoutPad * Cin;
    hipLaunchKernelGGL(k_wconv, grid1(n), dim3(256), 0, stream, w, o, Cout, CoutPad, Cin, taps);
  };
  auto pool = [&](const _Float16* in, _Float16* o, int H, int W, int C) {
    long long n = (long long)KB * (H / 2) * (W / 2) * C;
    hipLaunchKernelGGL(k_pool, grid1(n), dim3(256), 0, stream, in, o, H, W, C);
  };
  auto up2 = [&](const _Float16* in, _Float16* dst, int sH, int sW, int C, int dstC, int coff) {
    long long n = (long long)KB * 2 * sH * 2 * sW * C;
    hipLaunchKernelGGL(k_resize2x, grid1(n), dim3(256), 0, stream, in, dst, sH, sW, C, dstC, coff);
  };
  auto catc = [&](const _Float16* src, _Float16* dst, int H, int W, int Cs, int dstC, int coff) {
    long long n = (long long)KB * H * W * Cs;
    hipLaunchKernelGGL(k_copyc, grid1(n), dim3(256), 0, stream, src, dst, H, W, Cs, dstC, coff);
  };

#define CONV_LAUNCH(H_, W_, CIN_, COUTP_, COUT_, TAPS_, ACT_, PIXT_, inp, wptr, bptr, optr)   \
  {                                                                                           \
    long long waves_ = (long long)KB * (COUTP_ / 16) * ((H_ * W_) / (16 * PIXT_));            \
    hipLaunchKernelGGL((k_conv_wmma<H_, W_, CIN_, COUTP_, COUT_, TAPS_, ACT_, PIXT_>),        \
                       grid1(waves_ * 32), dim3(256), 0, stream, inp, wptr, bptr, optr);      \
  }

  // weight conversions
  wconv(ew2, w2f, 128, 128, 64, 25);
  wconv(ew3, w3f, 256, 256, 128, 25);
  wconv(ew4, w4f, 512, 512, 256, 25);
  wconv(cw1, wc1f, 256, 256, 512, 25);
  wconv(cw2, wc2f, 128, 128, 256, 1);
  wconv(dw1, wd1f, 256, 256, 768, 25);
  wconv(dw2, wd2f, 128, 128, 384, 25);
  wconv(dw3, wd3f, 64, 64, 192, 25);
  wconv(dwo, wdof, 4, 16, 64, 25);

  // encoder
  hipLaunchKernelGGL(k_norm, dim3(KB), dim3(256), 0, stream, img, xn);
  hipLaunchKernelGGL(k_conv1, grid1((long long)KB * 128 * 128), dim3(256), 0, stream,
                     xn, ew1, eb1, c1);
  pool(c1, f1, 128, 128, 64);
  CONV_LAUNCH(64, 64, 64, 128, 128, 25, 1, 4, f1, w2f, eb2, c2);
  pool(c2, f2, 64, 64, 128);
  CONV_LAUNCH(32, 32, 128, 256, 256, 25, 1, 4, f2, w3f, eb3, c3);
  pool(c3, f3, 32, 32, 256);
  CONV_LAUNCH(16, 16, 256, 512, 512, 25, 1, 4, f3, w4f, eb4, c4);
  pool(c4, f4, 16, 16, 512);
  CONV_LAUNCH(8, 8, 512, 256, 256, 25, 1, 4, f4, wc1f, cb1, g1);
  pool(g1, g1p, 8, 8, 256);
  CONV_LAUNCH(4, 4, 256, 128, 128, 1, 1, 1, g1p, wc2f, cb2, g2);

  // dense head -> affine
  hipLaunchKernelGGL(k_linear, grid1((long long)KB * 400), dim3(256), 0, stream,
                     g2, (const float*)nullptr, lw1, lb1, a1, 2048, 400, 1, 1);
  hipLaunchKernelGGL(k_linear, grid1((long long)KB * 200), dim3(256), 0, stream,
                     (const _Float16*)nullptr, a1, lw2, lb2, a2, 400, 200, 1, 0);
  hipLaunchKernelGGL(k_linear, grid1((long long)KB * 6), dim3(256), 0, stream,
                     (const _Float16*)nullptr, a2, lw3, lb3, aff, 200, 6, 2, 0);

  // decoder
  up2(f4, cat1, 8, 8, 512, 768, 0);
  catc(f3, cat1, 16, 16, 256, 768, 512);
  CONV_LAUNCH(16, 16, 768, 256, 256, 25, 1, 4, cat1, wd1f, db1, d1o);
  up2(d1o, cat2, 16, 16, 256, 384, 0);
  catc(f2, cat2, 32, 32, 128, 384, 256);
  CONV_LAUNCH(32, 32, 384, 128, 128, 25, 1, 4, cat2, wd2f, db2, d2o);
  up2(d2o, cat3, 32, 32, 128, 192, 0);
  catc(f1, cat3, 64, 64, 64, 192, 128);
  CONV_LAUNCH(64, 64, 192, 64, 64, 25, 1, 4, cat3, wd3f, db3, d3o);
  CONV_LAUNCH(64, 64, 64, 16, 4, 25, 2, 4, d3o, wdof, dbo, d4o);   // tanh
  hipLaunchKernelGGL(k_resize_disp, grid1((long long)KB * 4 * 128 * 128), dim3(256), 0, stream,
                     d4o, disp);

  // geometry
  hipLaunchKernelGGL(k_nodes, grid1((long long)KB * KV), dim3(256), 0, stream, aff, n1, n2);
  long long npix = (long long)KB * KIMG * KIMG;
  hipLaunchKernelGGL(k_render, grid1(npix), dim3(256), 0, stream, n1, out, 0); // m1_0
  hipLaunchKernelGGL(k_render, grid1(npix), dim3(256), 0, stream, n2, out, 2); // m2_0
  for (int it = 0; it < 3; ++it)
    hipLaunchKernelGGL(k_iter, grid1((long long)KB * KV), dim3(256), 0, stream, disp, n1, n2);
  hipLaunchKernelGGL(k_render, grid1(npix), dim3(256), 0, stream, n1, out, 1); // m1_T
  hipLaunchKernelGGL(k_render, grid1(npix), dim3(256), 0, stream, n2, out, 3); // m2_T

  (void)in_sizes; (void)n_in; (void)out_size; (void)ws_size;
}
